// RocketConv1d_36447092474162
// MI455X (gfx1250) — matrix operand order repaired and checked by reading, not compiled, not run on hardware
//
#include <hip/hip_runtime.h>

typedef _Float16 hlf;
typedef __attribute__((ext_vector_type(16))) _Float16 v16h;
typedef __attribute__((ext_vector_type(8)))  float    v8f;

#define KNUM    256          // number of ROCKET kernels
#define TLEN    2000         // series length
#define NTILES  (TLEN / 16)  // 125 time tiles (exact)
#define KSTEPS  4            // ceil(128/32) WMMA K-steps (taps zero-padded)
#define XW_LEN  2176         // >= TLEN-1 + 127 + 16 window, padded

__global__ __launch_bounds__(256)
void rocket_wmma_kernel(const float* __restrict__ x,
                        const float* __restrict__ w,
                        const float* __restrict__ bias,
                        float* __restrict__ out,
                        int Lmax, int pad)
{
    __shared__ hlf xw[XW_LEN];

    const int s    = blockIdx.x;       // series index (b*C + c), 256 blocks
    const int tid  = threadIdx.x;      // 0..255, 8 waves
    const int lane = tid & 31;
    const int wv   = tid >> 5;         // wave id 0..7
    const int mrow = lane & 15;        // M row within tile / N col within tile
    const int hlane = lane >> 4;       // 0: lanes 0-15, 1: lanes 16-31

    // ---- Stage x window into LDS as f16 with SAME-padding zeros baked in.
    //      xw[i] = x[s, i - pad], zero out of range.  y[t] uses xw[t + l].
    const float* xs = x + (size_t)s * TLEN;
    for (int i = tid; i < XW_LEN; i += 256) {
        int idx = i - pad;
        float v = (idx >= 0 && idx < TLEN) ? xs[idx] : 0.0f;
        xw[i] = (hlf)v;
    }

    // ---- Preload A (weight) fragments: 2 M-tiles x 4 K-steps, held in VGPRs
    //      for the whole time loop.  A layout used here (same k order as B):
    //      lane m-row, element i -> vgpr i/2 half i%2,
    //      k_local = hlane*16 + i
    v16h afrag[2][KSTEPS];
    float breg[2][8];
#pragma unroll
    for (int j = 0; j < 2; ++j) {
        const int mt  = wv * 2 + j;                  // M tile 0..15
        const int row = mt * 16 + mrow;              // kernel row for A gather
        const float* wr = w + (size_t)row * Lmax;
#pragma unroll
        for (int kk = 0; kk < KSTEPS; ++kk) {
            v16h a;
#pragma unroll
            for (int i = 0; i < 16; ++i) {
                int vg = i >> 1;
                int kl = hlane * 16 + i; (void)vg;
                int tap = kk * 32 + kl;
                a[i] = (hlf)((tap < Lmax) ? wr[tap] : 0.0f);
            }
            afrag[j][kk] = a;
        }
        // bias for the 8 C-rows this lane owns in tile j (rows hlane*8 + i)
#pragma unroll
        for (int i = 0; i < 8; ++i)
            breg[j][i] = bias[mt * 16 + hlane * 8 + i];
    }

    __syncthreads();

    // ---- Main loop over 125 time tiles: 8 WMMAs per tile per wave,
    //      fold bias/threshold/max immediately into running registers.
    float cnt[2][8], mx[2][8];
#pragma unroll
    for (int j = 0; j < 2; ++j)
#pragma unroll
        for (int i = 0; i < 8; ++i) { cnt[j][i] = 0.0f; mx[j][i] = -__builtin_inff(); }

    for (int nt = 0; nt < NTILES; ++nt) {
        const int tbase = nt * 16 + mrow + hlane * 16; // B gather base for this lane
        v8f acc0 = {}, acc1 = {};
#pragma unroll
        for (int kk = 0; kk < KSTEPS; ++kk) {
            // B fragment: 32x16 (K x N) f16; lanes 0-15 hold K=0..15 (col=lane),
            // lanes 16-31 hold K=16..31; element i -> k_local = hlane*16 + i.
            const hlf* bp = &xw[tbase + kk * 32];
            v16h b;
#pragma unroll
            for (int i = 0; i < 16; ++i) b[i] = bp[i];

            acc0 = __builtin_amdgcn_wmma_f32_16x16x32_f16(
                       false, afrag[0][kk], false, b, (short)0, acc0, false, false);
            acc1 = __builtin_amdgcn_wmma_f32_16x16x32_f16(
                       false, afrag[1][kk], false, b, (short)0, acc1, false, false);
        }
        // C/D layout: vgpr i = row (hlane? i+8 : i), lane%16 = time column.
#pragma unroll
        for (int i = 0; i < 8; ++i) {
            float v0 = acc0[i] + breg[0][i];
            cnt[0][i] += (v0 > 0.0f) ? 1.0f : 0.0f;
            mx[0][i]   = fmaxf(mx[0][i], v0);
            float v1 = acc1[i] + breg[1][i];
            cnt[1][i] += (v1 > 0.0f) ? 1.0f : 0.0f;
            mx[1][i]   = fmaxf(mx[1][i], v1);
        }
    }

    // ---- Reduce over the 16 time-columns (lanes 0-15 and 16-31 separately:
    //      xor masks 1,2,4,8 stay within each half-wave on wave32).
#pragma unroll
    for (int j = 0; j < 2; ++j)
#pragma unroll
        for (int i = 0; i < 8; ++i) {
            float c = cnt[j][i], m = mx[j][i];
#pragma unroll
            for (int d = 1; d < 16; d <<= 1) {
                c += __shfl_xor(c, d, 32);
                m  = fmaxf(m, __shfl_xor(m, d, 32));
            }
            cnt[j][i] = c; mx[j][i] = m;
        }

    // ---- Lanes 0 and 16 hold full reductions for rows (mt*16 + hlane*8 + i).
    //      Interleave [ppv, max] per kernel: out[s, 2k] , out[s, 2k+1].
    if (mrow == 0) {
        float* os = out + (size_t)s * (2 * KNUM);
#pragma unroll
        for (int j = 0; j < 2; ++j) {
            const int rbase = (wv * 2 + j) * 16 + hlane * 8;
#pragma unroll
            for (int i = 0; i < 8; ++i) {
                const int k = rbase + i;
                os[2 * k]     = cnt[j][i] * (1.0f / (float)TLEN);
                os[2 * k + 1] = mx[j][i];
            }
        }
    }
}

extern "C" void kernel_launch(void* const* d_in, const int* in_sizes, int n_in,
                              void* d_out, int out_size, void* d_ws, size_t ws_size,
                              hipStream_t stream) {
    (void)n_in; (void)out_size; (void)d_ws; (void)ws_size;
    const float* x    = (const float*)d_in[0];   // (8,1,32,2000) f32
    const float* w    = (const float*)d_in[1];   // (256, Lmax)   f32
    const float* bias = (const float*)d_in[2];   // (256,)        f32
    float* out = (float*)d_out;                  // (8,32,512)    f32

    const int Lmax = in_sizes[1] / KNUM;         // densified kernel length (odd, <=127)
    const int pad  = (Lmax - 1) / 2;             // SAME padding

    rocket_wmma_kernel<<<dim3(256), dim3(256), 0, stream>>>(x, w, bias, out, Lmax, pad);
}
